// FullModel_36043365548685
// MI455X (gfx1250) — matrix operand order repaired and checked by reading, not compiled, not run on hardware
//
#include <hip/hip_runtime.h>

#define LOOKBACK 20
#define TSTEPS   220
#define BATCH    1024

typedef unsigned short u16;
typedef unsigned int   u32;
typedef __attribute__((ext_vector_type(16))) __bf16 v16bf;
typedef __attribute__((ext_vector_type(8)))  float  v8f;
typedef __attribute__((ext_vector_type(4)))  float  f32x4;
typedef __attribute__((ext_vector_type(4)))  u32    u32x4;

// round-to-nearest-even f32 -> bf16 (bit pattern)
__device__ __forceinline__ u16 f2bf(float f) {
    u32 u = __float_as_uint(f);
    u32 r = u + 0x7FFFu + ((u >> 16) & 1u);
    return (u16)(r >> 16);
}
__device__ __forceinline__ float sigm(float x) {
    return 1.0f / (1.0f + __expf(-x));
}
__device__ __forceinline__ float tanh_f(float x) {
    float e = __expf(-2.0f * fabsf(x));
    float t = (1.0f - e) / (1.0f + e);
    return copysignf(t, x);
}

// A fragment (16x32 bf16, MxK) from LDS row-major [rows][96] bf16 tile.
// ISA layout: lane (half=lane/16, r=lane%16) element e -> M=r, K=(e/8)*16 + half*8 + (e%8)
__device__ __forceinline__ v16bf load_afrag(const u16* base, int row0, int kt, int hlf, int r) {
    union { v16bf v; u32x4 u[2]; } w;
    const u16* p = base + (row0 + r) * 96 + kt * 32 + hlf * 8;
    w.u[0] = *(const u32x4*)(p);        // K = kt*32 + hlf*8 + [0..8)
    w.u[1] = *(const u32x4*)(p + 16);   // K = kt*32 + 16 + hlf*8 + [0..8)
    return w.v;
}
// B fragment (32x16 bf16, KxN) pre-packed in global: [nt][kt][lane][16 halfs].
// Pointer is NOT __restrict__: the t-loop's stores to `out` may alias, which
// stops LICM from hoisting these loop-invariant loads (no scratch spills) while
// keeping them ordinary L2-cacheable global_load_b128s.
__device__ __forceinline__ v16bf load_bfrag(const u16* pack, int nt, int kt, int KT, int lane) {
    union { v16bf v; u32x4 u[2]; } w;
    const u32x4* p = (const u32x4*)(pack + (((nt * KT) + kt) * 32 + lane) * 16);
    w.u[0] = p[0];
    w.u[1] = p[1];
    return w.v;
}
__device__ __forceinline__ v8f wmma_bf16(v16bf a, v16bf b, v8f c) {
    return __builtin_amdgcn_wmma_f32_16x16x32_bf16(false, a, false, b, (short)0, c, false, false);
}

// Pre-pack a row-major f32 weight W[Nout][rowStride] into bf16 WMMA-B fragment order.
// B[k][n] = W[n][colOff + k]; dst idx = ((nt*KT+kt)*32 + lane)*16 + e
__global__ void prepack_wmmaB(const float* __restrict__ W, u16* __restrict__ dst,
                              int NT, int KT, int rowStride, int colOff) {
    int idx = blockIdx.x * blockDim.x + threadIdx.x;
    int total = NT * KT * 512;
    if (idx >= total) return;
    int e    = idx & 15;
    int lane = (idx >> 4) & 31;
    int tile = idx >> 9;
    int kt = tile % KT, nt = tile / KT;
    int n = nt * 16 + (lane & 15);
    int k = kt * 32 + (e >> 3) * 16 + (lane >> 4) * 8 + (e & 7);
    dst[idx] = f2bf(W[n * rowStride + colOff + k]);
}

// Persistent per-batch-tile kernel: 32 workgroups x 32 rows, 192 threads (6 wave32).
// Wave w owns output htile w (cols [16w,16w+16)) for BOTH row tiles (mt=0,1):
// each weight B-fragment is loaded once and reused by two WMMAs; A fragments are
// loaded once per (mt,kt) and reused across all four gates.
__global__ __launch_bounds__(192) __attribute__((amdgpu_waves_per_eu(4)))
void lstm_stack_kernel(const float* __restrict__ tact,
                       const float* __restrict__ act,
                       const float* __restrict__ b_ih1, const float* __restrict__ b_hh1,
                       const float* __restrict__ w_ih2, const float* __restrict__ w_hh2,
                       const float* __restrict__ b_ih2, const float* __restrict__ b_hh2,
                       const float* __restrict__ w_fc,  const float* __restrict__ b_fc,
                       const float* __restrict__ b_ih3, const float* __restrict__ b_hh3,
                       const u16* pw_ih1, const u16* pw_hh1,
                       const u16* pw_fc,
                       const u16* pw_ih3, const u16* pw_hh3,
                       float* out)
{
    __shared__ __align__(16) u16 x1s[32 * 96];        // tactile staging (t < LOOKBACK only)
    __shared__ __align__(16) u16 h1buf[2 * 32 * 96];  // h1 state, double-buffered
    __shared__ __align__(16) u16 o3s[32 * 96];        // fc output
    __shared__ __align__(16) u16 h3buf[2 * 32 * 96];  // h3 state, double-buffered
    __shared__ __align__(16) float h2s[32 * 4];       // h2 state (f32)
    __shared__ float bsum1[384], bsum3[384], bfcs[96], bsum2[16];
    __shared__ float wih2s[64], whh2s[64], wfc4[96 * 4];

    const int tid  = threadIdx.x;
    const int wg   = blockIdx.x;       // rows [wg*32, wg*32+32)
    const int wav  = tid >> 5;         // 0..5 == owned htile
    const int lane = tid & 31;
    const int hlf  = lane >> 4;
    const int nl   = lane & 15;

    // one-time preload
    for (int i = tid; i < 384; i += 192) { bsum1[i] = b_ih1[i] + b_hh1[i]; bsum3[i] = b_ih3[i] + b_hh3[i]; }
    for (int i = tid; i < 96;  i += 192) bfcs[i] = b_fc[i];
    if (tid < 16) bsum2[tid] = b_ih2[tid] + b_hh2[tid];
    if (tid < 64) { wih2s[tid] = w_ih2[tid]; whh2s[tid] = w_hh2[tid]; }
    for (int i = tid; i < 384; i += 192) wfc4[i] = w_fc[(i >> 2) * 100 + (i & 3)];  // w_fc[:, 0:4]
    for (int i = tid; i < 2 * 32 * 96; i += 192) { h1buf[i] = 0; h3buf[i] = 0; }
    if (tid < 128) h2s[tid] = 0.f;

    float c1st[2][8], c3st[2][8];
    #pragma unroll
    for (int m = 0; m < 2; ++m)
        #pragma unroll
        for (int g = 0; g < 8; ++g) { c1st[m][g] = 0.f; c3st[m][g] = 0.f; }
    float c2reg = 0.f;

    const int row2 = tid >> 2;  // LSTM2: thread -> (row, hidden j), tid < 128
    const int j2   = tid & 3;

    for (int t = 0; t < TSTEPS; ++t) {
        const u16* h1_rd = h1buf + (t & 1) * (32 * 96);
        u16*       h1_wr = h1buf + ((t + 1) & 1) * (32 * 96);
        const u16* h3_rd = h3buf + (t & 1) * (32 * 96);
        u16*       h3_wr = h3buf + ((t + 1) & 1) * (32 * 96);

        __syncthreads();  // B0: previous step's h3 write buffer is now our read buffer

        // stage0: stage tactiles for t<LOOKBACK (else LSTM1 reads h3_rd directly)
        if (t < LOOKBACK) {
            const float* tp = tact + ((size_t)t * BATCH + wg * 32) * 96;
            for (int i = tid; i < 32 * 96; i += 192) x1s[i] = f2bf(tp[i]);
        }

        // LSTM2 (hidden=4): pure VALU, one thread per (row, hidden unit)
        if (tid < 128) {
            const float* ap = act + ((size_t)t * BATCH + wg * 32 + row2) * 4;
            float x0 = ap[0], x1 = ap[1], x2 = ap[2], x3 = ap[3];
            float h0 = h2s[row2 * 4 + 0], h1 = h2s[row2 * 4 + 1];
            float h2v = h2s[row2 * 4 + 2], h3v = h2s[row2 * 4 + 3];
            float gg[4];
            #pragma unroll
            for (int ga = 0; ga < 4; ++ga) {
                int n = ga * 4 + j2;
                gg[ga] = bsum2[n]
                       + x0 * wih2s[n * 4 + 0] + x1 * wih2s[n * 4 + 1]
                       + x2 * wih2s[n * 4 + 2] + x3 * wih2s[n * 4 + 3]
                       + h0 * whh2s[n * 4 + 0] + h1 * whh2s[n * 4 + 1]
                       + h2v * whh2s[n * 4 + 2] + h3v * whh2s[n * 4 + 3];
            }
            c2reg = sigm(gg[1]) * c2reg + sigm(gg[0]) * tanh_f(gg[2]);
            h2s[row2 * 4 + j2] = sigm(gg[3]) * tanh_f(c2reg);  // same-wave rows: reads precede writes
        }
        __syncthreads();  // B1: x1s / h2s ready

        const u16* xsrc = (t < LOOKBACK) ? x1s : h3_rd;

        // ---- LSTM1: gates[mt][ga] for htile=wav; B frags shared across mt ----
        {
            v8f acc[2][4];
            #pragma unroll
            for (int ga = 0; ga < 4; ++ga) {
                const float b = bsum1[(ga * 6 + wav) * 16 + nl];
                acc[0][ga] = v8f{ b, b, b, b, b, b, b, b };
                acc[1][ga] = acc[0][ga];
            }
            #pragma unroll
            for (int kt = 0; kt < 3; ++kt) {
                v16bf ax0 = load_afrag(xsrc,  0,  kt, hlf, nl);
                v16bf ax1 = load_afrag(xsrc,  16, kt, hlf, nl);
                v16bf ah0 = load_afrag(h1_rd, 0,  kt, hlf, nl);
                v16bf ah1 = load_afrag(h1_rd, 16, kt, hlf, nl);
                #pragma unroll
                for (int ga = 0; ga < 4; ++ga) {
                    const int nt = ga * 6 + wav;
                    v16bf bih = load_bfrag(pw_ih1, nt, kt, 3, lane);
                    v16bf bhh = load_bfrag(pw_hh1, nt, kt, 3, lane);
                    acc[0][ga] = wmma_bf16(ax0, bih, acc[0][ga]);
                    acc[1][ga] = wmma_bf16(ax1, bih, acc[1][ga]);
                    acc[0][ga] = wmma_bf16(ah0, bhh, acc[0][ga]);
                    acc[1][ga] = wmma_bf16(ah1, bhh, acc[1][ga]);
                }
            }
            #pragma unroll
            for (int mt = 0; mt < 2; ++mt)
                #pragma unroll
                for (int g = 0; g < 8; ++g) {
                    float cn = sigm(acc[mt][1][g]) * c1st[mt][g]
                             + sigm(acc[mt][0][g]) * tanh_f(acc[mt][2][g]);
                    c1st[mt][g] = cn;
                    float hv = sigm(acc[mt][3][g]) * tanh_f(cn);
                    h1_wr[(mt * 16 + hlf * 8 + g) * 96 + wav * 16 + nl] = f2bf(hv);
                }
        }
        __syncthreads();  // B2: h1 (write buffer) ready

        // ---- FC: out3 = h1 @ Wfc[:,4:]^T (WMMA) + h2 @ Wfc[:,0:4]^T + b (rank-4 VALU) ----
        {
            const int n = wav * 16 + nl;
            const float w0 = wfc4[n * 4 + 0], w1 = wfc4[n * 4 + 1];
            const float w2 = wfc4[n * 4 + 2], w3 = wfc4[n * 4 + 3];
            const float bb = bfcs[n];
            v8f c[2];
            #pragma unroll
            for (int mt = 0; mt < 2; ++mt)
                #pragma unroll
                for (int g = 0; g < 8; ++g) {
                    const int M = mt * 16 + hlf * 8 + g;
                    f32x4 h2v = *(const f32x4*)&h2s[M * 4];
                    c[mt][g] = bb + h2v.x * w0 + h2v.y * w1 + h2v.z * w2 + h2v.w * w3;
                }
            #pragma unroll
            for (int kt = 0; kt < 3; ++kt) {
                v16bf a0 = load_afrag(h1_wr, 0,  kt, hlf, nl);
                v16bf a1 = load_afrag(h1_wr, 16, kt, hlf, nl);
                v16bf b  = load_bfrag(pw_fc, wav, kt, 3, lane);
                c[0] = wmma_bf16(a0, b, c[0]);
                c[1] = wmma_bf16(a1, b, c[1]);
            }
            #pragma unroll
            for (int mt = 0; mt < 2; ++mt)
                #pragma unroll
                for (int g = 0; g < 8; ++g)
                    o3s[(mt * 16 + hlf * 8 + g) * 96 + wav * 16 + nl] = f2bf(c[mt][g]);
        }
        __syncthreads();  // B3: out3 ready

        // ---- LSTM3: same structure; writes h3 + global output ----
        {
            v8f acc[2][4];
            #pragma unroll
            for (int ga = 0; ga < 4; ++ga) {
                const float b = bsum3[(ga * 6 + wav) * 16 + nl];
                acc[0][ga] = v8f{ b, b, b, b, b, b, b, b };
                acc[1][ga] = acc[0][ga];
            }
            #pragma unroll
            for (int kt = 0; kt < 3; ++kt) {
                v16bf ax0 = load_afrag(o3s,   0,  kt, hlf, nl);
                v16bf ax1 = load_afrag(o3s,   16, kt, hlf, nl);
                v16bf ah0 = load_afrag(h3_rd, 0,  kt, hlf, nl);
                v16bf ah1 = load_afrag(h3_rd, 16, kt, hlf, nl);
                #pragma unroll
                for (int ga = 0; ga < 4; ++ga) {
                    const int nt = ga * 6 + wav;
                    v16bf bih = load_bfrag(pw_ih3, nt, kt, 3, lane);
                    v16bf bhh = load_bfrag(pw_hh3, nt, kt, 3, lane);
                    acc[0][ga] = wmma_bf16(ax0, bih, acc[0][ga]);
                    acc[1][ga] = wmma_bf16(ax1, bih, acc[1][ga]);
                    acc[0][ga] = wmma_bf16(ah0, bhh, acc[0][ga]);
                    acc[1][ga] = wmma_bf16(ah1, bhh, acc[1][ga]);
                }
            }
            #pragma unroll
            for (int mt = 0; mt < 2; ++mt)
                #pragma unroll
                for (int g = 0; g < 8; ++g) {
                    float cn = sigm(acc[mt][1][g]) * c3st[mt][g]
                             + sigm(acc[mt][0][g]) * tanh_f(acc[mt][2][g]);
                    c3st[mt][g] = cn;
                    float hv = sigm(acc[mt][3][g]) * tanh_f(cn);
                    const int M = mt * 16 + hlf * 8 + g;
                    const int col = wav * 16 + nl;
                    h3_wr[M * 96 + col] = f2bf(hv);
                    if (t >= LOOKBACK)
                        out[((size_t)(t - LOOKBACK) * BATCH + wg * 32 + M) * 96 + col] = hv;
                }
        }
    }
}

extern "C" void kernel_launch(void* const* d_in, const int* in_sizes, int n_in,
                              void* d_out, int out_size, void* d_ws, size_t ws_size,
                              hipStream_t stream)
{
    (void)in_sizes; (void)n_in; (void)out_size; (void)ws_size;
    const float* tact  = (const float*)d_in[0];
    const float* act   = (const float*)d_in[1];
    const float* w_ih1 = (const float*)d_in[2];
    const float* w_hh1 = (const float*)d_in[3];
    const float* b_ih1 = (const float*)d_in[4];
    const float* b_hh1 = (const float*)d_in[5];
    const float* w_ih2 = (const float*)d_in[6];
    const float* w_hh2 = (const float*)d_in[7];
    const float* b_ih2 = (const float*)d_in[8];
    const float* b_hh2 = (const float*)d_in[9];
    const float* w_fc  = (const float*)d_in[10];
    const float* b_fc  = (const float*)d_in[11];
    const float* w_ih3 = (const float*)d_in[12];
    const float* w_hh3 = (const float*)d_in[13];
    const float* b_ih3 = (const float*)d_in[14];
    const float* b_hh3 = (const float*)d_in[15];

    // workspace layout (u16 elements): 4x 36864 LSTM weight packs + 9216 fc pack = 313,344 B
    u16* ws     = (u16*)d_ws;
    u16* pw_ih1 = ws +      0;
    u16* pw_hh1 = ws +  36864;
    u16* pw_ih3 = ws +  73728;
    u16* pw_hh3 = ws + 110592;
    u16* pw_fc  = ws + 147456;

    prepack_wmmaB<<<144, 256, 0, stream>>>(w_ih1, pw_ih1, 24, 3, 96, 0);
    prepack_wmmaB<<<144, 256, 0, stream>>>(w_hh1, pw_hh1, 24, 3, 96, 0);
    prepack_wmmaB<<<144, 256, 0, stream>>>(w_ih3, pw_ih3, 24, 3, 96, 0);
    prepack_wmmaB<<<144, 256, 0, stream>>>(w_hh3, pw_hh3, 24, 3, 96, 0);
    prepack_wmmaB<<< 36, 256, 0, stream>>>(w_fc,  pw_fc,   6, 3, 100, 4);

    lstm_stack_kernel<<<32, 192, 0, stream>>>(
        tact, act, b_ih1, b_hh1, w_ih2, w_hh2, b_ih2, b_hh2,
        w_fc, b_fc, b_ih3, b_hh3,
        pw_ih1, pw_hh1, pw_fc, pw_ih3, pw_hh3, (float*)d_out);
}
